// SwitchMoE_53987738910743
// MI455X (gfx1250) — matrix operand order repaired and checked by reading, not compiled, not run on hardware
//
#include <hip/hip_runtime.h>
#include <hip/hip_bf16.h>
#include <math.h>

#define N_TOK (8 * 2048)
#define DDIM  1024
#define EEXP  8
#define HDIM  256
#define LN_EPS 1e-5f
#define TILE_M 128
#define MAX_TILES (N_TOK / TILE_M)

// staged B slab: 256 N-rows x 32 K (bf16), padded row stride 40 ushorts (80 B)
#define STG_ROWS   256
#define STG_STRIDE 40
#define STG_ELEMS  (STG_ROWS * STG_STRIDE)

typedef __attribute__((ext_vector_type(16))) __bf16 v16bf;
typedef __attribute__((ext_vector_type(8)))  float  v8f;
typedef __attribute__((ext_vector_type(4)))  int    v4i;

typedef __attribute__((address_space(1))) v4i as1_v4i;
typedef __attribute__((address_space(3))) v4i as3_v4i;

union V16 { v16bf v; uint4 q[2]; };

__device__ __forceinline__ unsigned short f2bf(float f) {
  unsigned u = __float_as_uint(f);
  u += 0x7FFFu + ((u >> 16) & 1u);   // round-to-nearest-even
  return (unsigned short)(u >> 16);
}

__device__ __forceinline__ float gelu_exact(float v) {
  return 0.5f * v * (1.0f + erff(v * 0.70710678118654752f));
}

// ---------------- weight conversion / transpose to bf16 ----------------
// W1 [E,D,H] f32 -> w1t [E,H,D] bf16  (K-major for WMMA B operand of GEMM1)
__global__ __launch_bounds__(256) void convert_w1_kernel(
    const float* __restrict__ W1, unsigned short* __restrict__ w1t) {
  int i = blockIdx.x * 256 + threadIdx.x;      // over E*H*D
  int d = i % DDIM;
  int r = i / DDIM;
  int h = r % HDIM;
  int e = r / HDIM;
  w1t[i] = f2bf(W1[((size_t)e * DDIM + d) * HDIM + h]);
}

// W2 [E,H,D] f32 -> w2t [E,D,H] bf16  (K-major for WMMA B operand of GEMM2)
__global__ __launch_bounds__(256) void convert_w2_kernel(
    const float* __restrict__ W2, unsigned short* __restrict__ w2t) {
  int i = blockIdx.x * 256 + threadIdx.x;      // over E*D*H
  int h = i % HDIM;
  int r = i / HDIM;
  int d = r % DDIM;
  int e = r / DDIM;
  w2t[i] = f2bf(W2[((size_t)e * HDIM + h) * DDIM + d]);
}

// ---------------- gating + layernorm (one wave32 per token) ----------------
__global__ __launch_bounds__(256) void gate_ln_kernel(
    const float* __restrict__ x, const float* __restrict__ Wg,
    const float* __restrict__ ln_s, const float* __restrict__ ln_b,
    unsigned short* __restrict__ xin, int* __restrict__ expert_id,
    int* __restrict__ counts) {
  int tok  = (blockIdx.x * blockDim.x + threadIdx.x) >> 5;
  int lane = threadIdx.x & 31;
  if (tok >= N_TOK) return;
  const float* xr = x + (size_t)tok * DDIM;

  float4 xc[8];
  float red[10];
#pragma unroll
  for (int i = 0; i < 10; i++) red[i] = 0.f;

#pragma unroll
  for (int c = 0; c < 8; c++) {
    int idx = c * 128 + lane * 4;
    xc[c] = *(const float4*)(xr + idx);
    red[0] += xc[c].x + xc[c].y + xc[c].z + xc[c].w;
    red[1] += xc[c].x * xc[c].x + xc[c].y * xc[c].y +
              xc[c].z * xc[c].z + xc[c].w * xc[c].w;
#pragma unroll
    for (int e = 0; e < EEXP; e++) {
      float4 w = *(const float4*)(Wg + (size_t)e * DDIM + idx);
      red[2 + e] += xc[c].x * w.x + xc[c].y * w.y + xc[c].z * w.z + xc[c].w * w.w;
    }
  }
#pragma unroll
  for (int i = 0; i < 10; i++)
#pragma unroll
    for (int off = 16; off; off >>= 1)
      red[i] += __shfl_xor(red[i], off, 32);

  int be = 0; float bv = red[2];
#pragma unroll
  for (int e = 1; e < EEXP; e++)
    if (red[2 + e] > bv) { bv = red[2 + e]; be = e; }   // strict > == first argmax

  float mu   = red[0] * (1.0f / DDIM);
  float var  = red[1] * (1.0f / DDIM) - mu * mu;
  float rstd = rsqrtf(var + LN_EPS);

  unsigned short* xo = xin + (size_t)tok * DDIM;
  const float* sr = ln_s + (size_t)be * DDIM;
  const float* br = ln_b + (size_t)be * DDIM;
#pragma unroll
  for (int c = 0; c < 8; c++) {
    int idx = c * 128 + lane * 4;
    float4 s4 = *(const float4*)(sr + idx);
    float4 b4 = *(const float4*)(br + idx);
    ushort4 o;
    o.x = f2bf((xc[c].x - mu) * rstd * s4.x + b4.x);
    o.y = f2bf((xc[c].y - mu) * rstd * s4.y + b4.y);
    o.z = f2bf((xc[c].z - mu) * rstd * s4.z + b4.z);
    o.w = f2bf((xc[c].w - mu) * rstd * s4.w + b4.w);
    *(ushort4*)(xo + idx) = o;
  }
  if (lane == 0) {
    expert_id[tok] = be;
    atomicAdd(&counts[be], 1);
  }
}

// ---------------- tiny routing kernels ----------------
__global__ void zero_counts_kernel(int* counts) {
  if (threadIdx.x < EEXP) counts[threadIdx.x] = 0;
}

__global__ void scan_kernel(const int* __restrict__ counts,
                            int* __restrict__ offsets, int* __restrict__ cursor) {
  if (threadIdx.x == 0 && blockIdx.x == 0) {
    int o = 0;
    for (int e = 0; e < EEXP; e++) {
      offsets[e] = o; cursor[e] = o; o += counts[e];
    }
  }
}

__global__ __launch_bounds__(256) void scatter_kernel(
    const int* __restrict__ expert_id, int* __restrict__ cursor,
    int* __restrict__ order) {
  int t = blockIdx.x * 256 + threadIdx.x;
  if (t < N_TOK) {
    int e = expert_id[t];
    int p = atomicAdd(&cursor[e], 1);
    order[p] = t;
  }
}

// -------- async stage of one B k-slab (256 N-rows x 32 K bf16) into LDS ------
// wb: bf16 weight base (K-major rows of length kdim), rows [rowbase, rowbase+256)
// Every thread issues 4 x async-b128; one slab = 16 KB moved with 32
// wave-instructions per block, tracked on ASYNCcnt.
__device__ __forceinline__ void stage_slab(const unsigned short* wb, int kdim,
                                           int rowbase, int k0,
                                           unsigned short* buf, int tid) {
#pragma unroll
  for (int i = 0; i < 4; i++) {
    int chunk = tid + i * 256;          // 0..1023
    int n = chunk >> 2, part = chunk & 3;
    const unsigned short* g = wb + (size_t)(rowbase + n) * kdim + k0 + part * 8;
    unsigned short* l = buf + n * STG_STRIDE + part * 8;
    __builtin_amdgcn_global_load_async_to_lds_b128(
        (as1_v4i*)(unsigned long long)g, (as3_v4i*)(unsigned long long)l, 0, 0);
  }
}

// ---------------- fused expert FFN ----------------
// block = 256 threads = 8 waves; tile = 128 tokens of one expert.
// wave w owns rows [w*16, w*16+16) and the full N extent; the B operand
// k-slab is staged once in LDS (double buffered, async) and shared by all
// 8 waves -> no redundant weight traffic inside a block.
__global__ __launch_bounds__(256) void ffn_kernel(
    const unsigned short* __restrict__ xin,   // [N, D] bf16
    const unsigned short* __restrict__ w1t,   // [E, H, D] bf16 (K-major)
    const unsigned short* __restrict__ w2t,   // [E, D, H] bf16 (K-major)
    const float* __restrict__ b1,             // [E, H]
    const float* __restrict__ b2,             // [E, D]
    const int* __restrict__ counts, const int* __restrict__ offsets,
    const int* __restrict__ order, float* __restrict__ out) {
  __shared__ int toks[TILE_M];
  __shared__ unsigned short hsh[TILE_M][HDIM + 16];   // bf16 h tile, padded
  __shared__ unsigned short stg[2][STG_ELEMS];        // double-buffered B slab

  int e    = blockIdx.x / MAX_TILES;
  int tile = blockIdx.x % MAX_TILES;
  int cnt  = counts[e];
  int mbase = tile * TILE_M;
  if (mbase >= cnt) return;
  int base = offsets[e];

  int tid = threadIdx.x;
  if (tid < TILE_M) {
    int r = mbase + tid;
    toks[tid] = order[base + (r < cnt ? r : 0)];
  }

  int wave = tid >> 5, lane = tid & 31;
  int lo = lane & 15;
  bool hi = lane >= 16;
  int kA = hi ? 8 : 0;    // A-operand chunk offset (16-bit A layout)
  int kB = hi ? 8 : 0;    // B-operand chunk offset (16-bit B layout)

  const unsigned short* wb1 = w1t + (size_t)e * HDIM * DDIM;
  const unsigned short* wb2 = w2t + (size_t)e * DDIM * HDIM;

  v8f zero = {0.f, 0.f, 0.f, 0.f, 0.f, 0.f, 0.f, 0.f};
  v8f acc[16];

  // prefetch first W1 slab, sync once so toks[] is also visible
  stage_slab(wb1, DDIM, 0, 0, &stg[0][0], tid);
  __syncthreads();

  // ---- GEMM1: h[128x256] = xin_tile[128x1024] @ W1_e[1024x256] ----
  const unsigned short* arow = xin + (size_t)toks[wave * 16 + lo] * DDIM;
#pragma unroll
  for (int nt = 0; nt < 16; nt++) acc[nt] = zero;

  for (int ks = 0; ks < 32; ks++) {
    __builtin_amdgcn_s_wait_asynccnt(0);
    __syncthreads();
    if (ks + 1 < 32)
      stage_slab(wb1, DDIM, 0, (ks + 1) * 32, &stg[(ks + 1) & 1][0], tid);

    int k0 = ks * 32;
    const unsigned short* bs = &stg[ks & 1][0];
    V16 a;
    a.q[0] = *(const uint4*)(arow + k0 + kA);
    a.q[1] = *(const uint4*)(arow + k0 + 16 + kA);
#pragma unroll
    for (int nt = 0; nt < 16; nt++) {
      const unsigned short* bp = bs + (nt * 16 + lo) * STG_STRIDE + kB;
      V16 b;
      b.q[0] = *(const uint4*)(bp);
      b.q[1] = *(const uint4*)(bp + 16);
      acc[nt] = __builtin_amdgcn_wmma_f32_16x16x32_bf16(
          false, a.v, false, b.v, (short)0, acc[nt], false, false);
    }
  }

  // bias + exact GELU -> bf16 h in LDS (each wave writes only its own rows)
#pragma unroll
  for (int nt = 0; nt < 16; nt++) {
    int n = nt * 16 + lo;
    float b1v = b1[(size_t)e * HDIM + n];
#pragma unroll
    for (int j = 0; j < 8; j++) {
      float v = acc[nt][j] + b1v;
      hsh[wave * 16 + j + (hi ? 8 : 0)][n] = f2bf(gelu_exact(v));
    }
  }
  __syncthreads();   // all waves done with GEMM1 stage buffers

  // ---- GEMM2: y[128x1024] = h[128x256] @ W2_e[256x1024] ----
  // 32 slabs: c-chunk = s>>3 (256 output cols), ks = s&7 (K offset)
  stage_slab(wb2, HDIM, 0, 0, &stg[0][0], tid);

  const unsigned short* ap = &hsh[wave * 16 + lo][0];
  for (int s = 0; s < 32; s++) {
    int c = s >> 3, ks = s & 7;
    __builtin_amdgcn_s_wait_asynccnt(0);
    __syncthreads();
    if (s + 1 < 32) {
      int c2 = (s + 1) >> 3, ks2 = (s + 1) & 7;
      stage_slab(wb2, HDIM, c2 * 256, ks2 * 32, &stg[(s + 1) & 1][0], tid);
    }
    if (ks == 0) {
#pragma unroll
      for (int nt = 0; nt < 16; nt++) acc[nt] = zero;
    }

    int k0 = ks * 32;
    const unsigned short* bs = &stg[s & 1][0];
    V16 a;
    a.q[0] = *(const uint4*)(ap + k0 + kA);
    a.q[1] = *(const uint4*)(ap + k0 + 16 + kA);
#pragma unroll
    for (int nt = 0; nt < 16; nt++) {
      const unsigned short* bp = bs + (nt * 16 + lo) * STG_STRIDE + kB;
      V16 b;
      b.q[0] = *(const uint4*)(bp);
      b.q[1] = *(const uint4*)(bp + 16);
      acc[nt] = __builtin_amdgcn_wmma_f32_16x16x32_bf16(
          false, a.v, false, b.v, (short)0, acc[nt], false, false);
    }

    if (ks == 7) {    // chunk done: bias + scatter fp32 rows
#pragma unroll
      for (int nt = 0; nt < 16; nt++) {
        int n = c * 256 + nt * 16 + lo;
        float b2v = b2[(size_t)e * DDIM + n];
#pragma unroll
        for (int j = 0; j < 8; j++) {
          int m = wave * 16 + j + (hi ? 8 : 0);
          if (mbase + m < cnt)
            out[(size_t)toks[m] * DDIM + n] = acc[nt][j] + b2v;
        }
      }
    }
  }
}

// ---------------- host-side launcher ----------------
extern "C" void kernel_launch(void* const* d_in, const int* in_sizes, int n_in,
                              void* d_out, int out_size, void* d_ws, size_t ws_size,
                              hipStream_t stream) {
  const float* x    = (const float*)d_in[0];
  const float* Wg   = (const float*)d_in[1];
  const float* ln_s = (const float*)d_in[2];
  const float* ln_b = (const float*)d_in[3];
  const float* W1   = (const float*)d_in[4];
  const float* b1   = (const float*)d_in[5];
  const float* W2   = (const float*)d_in[6];
  const float* b2   = (const float*)d_in[7];
  float* out = (float*)d_out;

  char* ws = (char*)d_ws;
  size_t off = 0;
  auto carve = [&](size_t bytes) -> char* {
    char* p = ws + off;
    off += (bytes + 255) & ~(size_t)255;
    return p;
  };
  unsigned short* xin = (unsigned short*)carve((size_t)N_TOK * DDIM * 2);
  unsigned short* w1t = (unsigned short*)carve((size_t)EEXP * HDIM * DDIM * 2);
  unsigned short* w2t = (unsigned short*)carve((size_t)EEXP * DDIM * HDIM * 2);
  int* expert_id = (int*)carve((size_t)N_TOK * 4);
  int* order     = (int*)carve((size_t)N_TOK * 4);
  int* counts    = (int*)carve(256);
  int* offsets   = (int*)carve(256);
  int* cursor    = (int*)carve(256);

  const int wconv_blocks = (EEXP * HDIM * DDIM) / 256;   // 8192
  convert_w1_kernel<<<wconv_blocks, 256, 0, stream>>>(W1, w1t);
  convert_w2_kernel<<<wconv_blocks, 256, 0, stream>>>(W2, w2t);

  zero_counts_kernel<<<1, 32, 0, stream>>>(counts);

  gate_ln_kernel<<<N_TOK / 8, 256, 0, stream>>>(x, Wg, ln_s, ln_b, xin,
                                                expert_id, counts);

  scan_kernel<<<1, 1, 0, stream>>>(counts, offsets, cursor);

  scatter_kernel<<<N_TOK / 256, 256, 0, stream>>>(expert_id, cursor, order);

  ffn_kernel<<<EEXP * MAX_TILES, 256, 0, stream>>>(
      xin, w1t, w2t, b1, b2, counts, offsets, order, out);
}
